// MonotonicAttention_29618094473429
// MI455X (gfx1250) — matrix operand order repaired and checked by reading, not compiled, not run on hardware
//
#include <hip/hip_runtime.h>

// ---------------------------------------------------------------------------
// MonotonicAttention: out = softmax(causal_mask(Q K^T / sqrt(64)))  (V unused)
// B=16, S=2048, D=64. Output (B,S,S) f32 = 268 MB -> HBM-write-bound (~12us
// floor @ 23.3 TB/s). One-pass design: scores live in LDS, single HBM write.
// ---------------------------------------------------------------------------

#define S_LEN   2048
#define D_DIM   64
#define N_BATCH 16
#define QBLK    16          // query rows per workgroup (one WMMA M tile)
#define WAVES   8
#define THREADS 256
#define ROWPAD  2068        // LDS row stride in floats (2048 + 20 pad, 16B mult)

typedef __attribute__((ext_vector_type(16))) __bf16 v16bf;
typedef __attribute__((ext_vector_type(8)))  float  v8f;
typedef __attribute__((ext_vector_type(4)))  float  v4f;

__device__ __forceinline__ unsigned f_as_u(float x) {
    union { float f; unsigned u; } c; c.f = x; return c.u;
}

// Pack 16 f32 (linear in f[0..3]) into a bf16x16 WMMA fragment, 1 v_perm per
// bf16 pair: dst = { hi.bf16, lo.bf16 } via byte select {7,6,3,2}.
__device__ __forceinline__ v16bf pack_bf16x16(const v4f f[4]) {
    union { v16bf bf; unsigned u[8]; } r;
    #pragma unroll
    for (int d = 0; d < 8; ++d) {
        unsigned lo = f_as_u(f[d >> 1][(d & 1) * 2]);
        unsigned hi = f_as_u(f[d >> 1][(d & 1) * 2 + 1]);
        r.u[d] = __builtin_amdgcn_perm(hi, lo, 0x07060302u);
    }
    return r.bf;
}

__global__ __launch_bounds__(THREADS)
void monotonic_attn_kernel(const float* __restrict__ Q,
                           const float* __restrict__ K,
                           float* __restrict__ out)
{
    extern __shared__ float smem[];              // [QBLK][ROWPAD]

    const int wg   = blockIdx.x;                 // 0 .. B*(S/16)-1
    const int b    = wg >> 7;                    // / (S/16)
    const int qb   = wg & 127;
    const int q0   = qb * QBLK;

    const int tid  = threadIdx.x;
    const int wave = tid >> 5;
    const int lane = tid & 31;
    const int half = lane >> 4;                  // 0 or 1
    const int l15  = lane & 15;

    // -------------------- Phase 1: scores via bf16 WMMA --------------------
    // A fragment (Q row m = l15): element e -> K-dim half*8+e (e<8),
    // 16+half*8+e-8 (e>=8). Linear float4 coverage: q4[half*2 .. half*2+1],
    // q4[4+half*2 .. 5+half*2] for k-step 0; +8 float4 for k-step 1.
    // 1/sqrt(dk) is folded into the A-fragment conversion.
    const float* qrow = Q + ((size_t)b * S_LEN + (size_t)(q0 + l15)) * D_DIM;
    const v4f*   q4p  = (const v4f*)qrow;
    v16bf a0, a1;
    {
        v4f f0[4], f1[4];
        #pragma unroll
        for (int i = 0; i < 2; ++i) {
            f0[i]     = q4p[half * 2 + i]     * 0.125f;
            f0[2 + i] = q4p[4 + half * 2 + i] * 0.125f;
            f1[i]     = q4p[8 + half * 2 + i]  * 0.125f;
            f1[2 + i] = q4p[12 + half * 2 + i] * 0.125f;
        }
        a0 = pack_bf16x16(f0);
        a1 = pack_bf16x16(f1);
    }

    const int ntiles = qb + 1;                   // causal: only tiles k0 <= q0
    const int ncols  = ntiles * 16;              // == q0 + 16

    for (int t = wave; t < ntiles; t += WAVES) { // wave-uniform branch (EXEC=~0)
        const int k0 = t * 16;
        // B fragment: element e -> K[k0+n][half*8+e] (e<8), [16+half*8+e-8] (e>=8) (k-step 0), +32 (step 1)
        const float* krow = K + ((size_t)b * S_LEN + (size_t)(k0 + l15)) * D_DIM;
        const v4f*   k4p  = (const v4f*)krow;
        v4f kf0[4], kf1[4];
        #pragma unroll
        for (int i = 0; i < 4; ++i) {
            kf0[i] = k4p[half * 2 + (i & 1) + (i >> 1) * 4];      // d = half*8..+7, 16+half*8..+7
            kf1[i] = k4p[8 + half * 2 + (i & 1) + (i >> 1) * 4];  // d = 32+ same
        }
        const v16bf b0 = pack_bf16x16(kf0);
        const v16bf b1 = pack_bf16x16(kf1);

        v8f c = {};
        c = __builtin_amdgcn_wmma_f32_16x16x32_bf16(false, a0, false, b0,
                                                    (short)0, c, false, false);
        c = __builtin_amdgcn_wmma_f32_16x16x32_bf16(false, a1, false, b1,
                                                    (short)0, c, false, false);

        // C layout: VGPR r -> row m = r + 8*half, col n = l15.
        float* dst = smem + (size_t)(k0 + l15);
        if (t == qb) {                           // diagonal tile: apply mask
            #pragma unroll
            for (int r = 0; r < 8; ++r) {
                const int m = r + 8 * half;
                float v = c[r];
                if (l15 > m) v = -1.0e8f;        // k = q0+l15 > q0+m
                dst[m * ROWPAD] = v;
            }
        } else {                                 // strictly below diagonal
            #pragma unroll
            for (int r = 0; r < 8; ++r)
                dst[(r + 8 * half) * ROWPAD] = c[r];
        }
    }
    __syncthreads();

    // -------------------- Phase 2: row softmax + single HBM write ----------
    const int nvec = ncols >> 2;                 // float4 count (mult of 4)
    #pragma unroll
    for (int rr = 0; rr < 2; ++rr) {
        const int m = wave * 2 + rr;             // 8 waves x 2 rows = 16 rows
        const int q = q0 + m;
        v4f* row4 = (v4f*)(smem + m * ROWPAD);

        // row max over valid region (vectorized LDS reads)
        float mx = -1.0e30f;
        for (int i = lane; i < nvec; i += 32) {
            v4f v = row4[i];
            mx = fmaxf(mx, fmaxf(fmaxf(v[0], v[1]), fmaxf(v[2], v[3])));
        }
        #pragma unroll
        for (int o = 16; o > 0; o >>= 1) mx = fmaxf(mx, __shfl_xor(mx, o, 32));

        // sum of exp; store exp back to LDS
        float sum = 0.0f;
        for (int i = lane; i < nvec; i += 32) {
            v4f v = row4[i];
            v4f e = { __expf(v[0] - mx), __expf(v[1] - mx),
                      __expf(v[2] - mx), __expf(v[3] - mx) };
            row4[i] = e;                         // masked -1e8 -> exactly 0
            sum += (e[0] + e[1]) + (e[2] + e[3]);
        }
        #pragma unroll
        for (int o = 16; o > 0; o >>= 1) sum += __shfl_xor(sum, o, 32);
        const float inv = __builtin_amdgcn_rcpf(sum);

        // coalesced float4 stores: valid region scaled, causal tail = 0
        v4f* orow4 = (v4f*)(out + ((size_t)b * S_LEN + (size_t)q) * S_LEN);
        for (int i = lane; i < nvec; i += 32)
            orow4[i] = row4[i] * inv;
        const v4f z = { 0.0f, 0.0f, 0.0f, 0.0f };
        for (int i = nvec + lane; i < (S_LEN >> 2); i += 32)
            orow4[i] = z;
    }
}

extern "C" void kernel_launch(void* const* d_in, const int* in_sizes, int n_in,
                              void* d_out, int out_size, void* d_ws, size_t ws_size,
                              hipStream_t stream) {
    const float* Q = (const float*)d_in[0];
    const float* K = (const float*)d_in[1];
    // d_in[2] = V (unused by reference output), d_in[3] = mask (recomputed
    // analytically as strict-upper-triangular causal), d_in[4] = dk (=64)
    float* out = (float*)d_out;

    const size_t lds_bytes = (size_t)QBLK * ROWPAD * sizeof(float); // ~129.3 KB
    (void)hipFuncSetAttribute((const void*)monotonic_attn_kernel,
                              hipFuncAttributeMaxDynamicSharedMemorySize,
                              (int)lds_bytes);

    const int nblocks = N_BATCH * (S_LEN / QBLK);  // 16 * 128 = 2048 WGs
    monotonic_attn_kernel<<<dim3(nblocks), dim3(THREADS), lds_bytes, stream>>>(Q, K, out);
}
